// HGT_13494787244588
// MI455X (gfx1250) — matrix operand order repaired and checked by reading, not compiled, not run on hardware
//
#include <hip/hip_runtime.h>
#include <math.h>

// ---------------------------------------------------------------------------
// Types for CDNA5 WMMA (wave32)
// ---------------------------------------------------------------------------
typedef __attribute__((ext_vector_type(16))) __bf16 v16bf;
typedef __attribute__((ext_vector_type(8)))  float  v8f;

union BfTile {
    v16bf          v;
    unsigned       u[8];
    uint4          q4[2];
    unsigned short s[16];
};

#define NI 40000
#define NV 32
#define FDIM 128
#define HEADS 8
#define DH 16
#define EII 640000
#define EIV 40000
#define EVI 40000
#define NLAYER 4
#define SCALE_ATT 0.25f  // 1/sqrt(16)

// ---------------------------------------------------------------------------
// Small device helpers
// ---------------------------------------------------------------------------
__device__ __forceinline__ unsigned short f2bf(float f) {
    unsigned u = __float_as_uint(f);
    u += 0x7FFFu + ((u >> 16) & 1u);   // round-to-nearest-even
    return (unsigned short)(u >> 16);
}
__device__ __forceinline__ unsigned fkey(float f) {  // monotone u32 key for float max
    unsigned u = __float_as_uint(f);
    return (u & 0x80000000u) ? ~u : (u | 0x80000000u);
}
__device__ __forceinline__ float funkey(unsigned u) {
    unsigned v = (u & 0x80000000u) ? (u & 0x7FFFFFFFu) : ~u;
    return __uint_as_float(v);
}
__device__ __forceinline__ float gelu_exact(float x) {
    return 0.5f * x * (1.0f + erff(x * 0.70710678118654752f));
}

// ---------------------------------------------------------------------------
// Elementwise / conversion kernels
// ---------------------------------------------------------------------------
__global__ void k_fill_u32(unsigned* p, unsigned val, int n) {
    int i = blockIdx.x * blockDim.x + threadIdx.x;
    if (i < n) p[i] = val;
}
__global__ void k_copy_f32(float* dst, const float* src, int n) {
    int i = blockIdx.x * blockDim.x + threadIdx.x;
    if (i < n) dst[i] = src[i];
}
__global__ void k_f32_to_bf16(unsigned short* dst, const float* src, int n) {
    int i = blockIdx.x * blockDim.x + threadIdx.x;
    if (i < n) dst[i] = f2bf(src[i]);
}
__global__ void k_gelu_bf16(unsigned short* dst, const float* src, int n) {
    int i = blockIdx.x * blockDim.x + threadIdx.x;
    if (i < n) dst[i] = f2bf(gelu_exact(src[i]));
}
// dst[r*width+c] = src[r*stride + col0 + c]
__global__ void k_slice_f32(float* dst, const float* src, int rows, int width,
                            int stride, int col0) {
    int i = blockIdx.x * blockDim.x + threadIdx.x;
    if (i >= rows * width) return;
    int r = i / width, c = i - r * width;
    dst[i] = src[(size_t)r * stride + col0 + c];
}
__global__ void k_slice_bf16(unsigned short* dst, const float* src, int rows,
                             int width, int stride, int col0) {
    int i = blockIdx.x * blockDim.x + threadIdx.x;
    if (i >= rows * width) return;
    int r = i / width, c = i - r * width;
    dst[i] = f2bf(src[(size_t)r * stride + col0 + c]);
}

// ---------------------------------------------------------------------------
// Weight conversion into WMMA B-operand packed layout.
// Packed layout: per K-tile kt (32 rows of K) and M-tile mt (16 cols),
//   dst[((kt*tn + mt)*32 + lane)*16 + i]  = B[kt*32 + (lane>=16?16:0) + i,
//                                             mt*16 + (lane&15)]
// so each lane's 16 bf16 B-register elements are contiguous (2x b128 loads).
// transposed=1 means src is torch-style [M,K] used as x @ W.T
// ---------------------------------------------------------------------------
__global__ void k_wconv_pack(unsigned short* dst, const float* src, int K, int M,
                             int transposed) {
    int idx = blockIdx.x * blockDim.x + threadIdx.x;
    if (idx >= K * M) return;
    int tn   = M >> 4;
    int i    = idx & 15;
    int lane = (idx >> 4) & 31;
    int tile = idx >> 9;
    int kt = tile / tn, mt = tile - kt * tn;
    int kk = kt * 32 + ((lane >> 4) << 4) + i;
    int mm = mt * 16 + (lane & 15);
    float v = transposed ? src[(size_t)mm * K + kk] : src[(size_t)kk * M + mm];
    dst[idx] = f2bf(v);
}
// Block-diagonal 128x128 (H diagonal 16x16 blocks) directly in packed layout
__global__ void k_wblockdiag_pack(unsigned short* dst, const float* src) {
    int idx = blockIdx.x * blockDim.x + threadIdx.x;
    if (idx >= FDIM * FDIM) return;
    const int tn = FDIM >> 4;
    int i    = idx & 15;
    int lane = (idx >> 4) & 31;
    int tile = idx >> 9;
    int kt = tile / tn, mt = tile - kt * tn;
    int kk = kt * 32 + ((lane >> 4) << 4) + i;
    int mm = mt * 16 + (lane & 15);
    float v = ((kk >> 4) == (mm >> 4))
                  ? src[(kk >> 4) * 256 + (kk & 15) * 16 + (mm & 15)]
                  : 0.0f;
    dst[idx] = f2bf(v);
}

// ---------------------------------------------------------------------------
// bf16 WMMA GEMM:  C[N,M] (f32) = A[N,K] (bf16, row-major) * B (packed) + bias
// K is a compile-time constant (32/64/128) -> fully unrolled WMMA chain.
// One wave per 16x16 output tile; requires N%16==0, M%16==0.
// ---------------------------------------------------------------------------
template <int K>
__global__ void k_gemm_bf16_t(const unsigned short* __restrict__ A,
                              const unsigned short* __restrict__ Bp,
                              const float* __restrict__ bias,
                              float* __restrict__ C, int N, int M) {
    int tn    = M >> 4;
    int total = (N >> 4) * tn;
    int w     = (int)((blockIdx.x * blockDim.x + threadIdx.x) >> 5);
    if (w >= total) return;                       // wave-uniform: EXEC stays full
    int lane    = threadIdx.x & 31;
    int rowTile = w / tn, colTile = w - rowTile * tn;
    int n0 = rowTile << 4, m0 = colTile << 4;
    int hi   = lane >> 4;                         // K-half select
    int arow = n0 + (lane & 15);
    int bcol = m0 + (lane & 15);
    const unsigned* A32 = (const unsigned*)(A + (size_t)arow * K);

    v8f acc = {};
#pragma unroll
    for (int kt = 0; kt < K / 32; ++kt) {
        BfTile a, b;
        int ka2 = kt * 16 + (hi << 3);            // dword-pair base index
        a.q4[0] = *(const uint4*)(A32 + ka2);       // K = kb + {0..7}  (+16 if hi)
        a.q4[1] = *(const uint4*)(A32 + ka2 + 4);   // K = kb + {8..15} (+16 if hi)
        const unsigned* bp =
            (const unsigned*)Bp + ((size_t)(kt * tn + colTile) * 32 + lane) * 8;
        b.q4[0] = *(const uint4*)(bp);
        b.q4[1] = *(const uint4*)(bp + 4);
        acc = __builtin_amdgcn_wmma_f32_16x16x32_bf16(
            false, a.v, false, b.v, (short)0, acc, false, false);
    }
    float bv = bias ? bias[bcol] : 0.0f;
    int r0 = n0 + (hi << 3);
#pragma unroll
    for (int i = 0; i < 8; ++i)
        C[(size_t)(r0 + i) * M + bcol] = acc[i] + bv;
}

// ---------------------------------------------------------------------------
// Edge attention kernels (joint segment softmax per destination node)
// ---------------------------------------------------------------------------
// Pass 1: score + running per-(dst,head) max via monotone-key u32 atomicMax
__global__ void k_edge_score(const float* __restrict__ q,   // [Nd,128] dst-type q
                             const float* __restrict__ kr,  // [Ns,128] rel-xformed k
                             const int* __restrict__ src, const int* __restrict__ dst,
                             const float* __restrict__ prel,  // [H]
                             int E, float* __restrict__ sc, unsigned* __restrict__ mx) {
    int idx = blockIdx.x * blockDim.x + threadIdx.x;
    if (idx >= E * HEADS) return;
    int e = idx >> 3, h = idx & 7;
    int s = src[e], d = dst[e];
    const float4* qp = (const float4*)(q  + (size_t)d * FDIM + h * DH);
    const float4* kp = (const float4*)(kr + (size_t)s * FDIM + h * DH);
    float acc = 0.0f;
#pragma unroll
    for (int j = 0; j < 4; ++j) {
        float4 a = qp[j], b = kp[j];
        acc += a.x * b.x + a.y * b.y + a.z * b.z + a.w * b.w;
    }
    acc *= prel[h] * SCALE_ATT;
    sc[(size_t)e * HEADS + h] = acc;
    atomicMax(mx + (size_t)d * HEADS + h, fkey(acc));
}
// Pass 2: unnormalized exp weights -> denom + weighted message accumulation
__global__ void k_edge_accum(const float* __restrict__ sc, const unsigned* __restrict__ mx,
                             const int* __restrict__ src, const int* __restrict__ dst,
                             const float* __restrict__ vr,  // [Ns,128]
                             float* __restrict__ den, float* __restrict__ agg, int E) {
    int idx = blockIdx.x * blockDim.x + threadIdx.x;
    if (idx >= E * HEADS) return;
    int e = idx >> 3, h = idx & 7;
    int s = src[e], d = dst[e];
    float m = funkey(mx[(size_t)d * HEADS + h]);
    if (!isfinite(m)) m = 0.0f;
    float ex = __expf(sc[(size_t)e * HEADS + h] - m);
    atomicAdd(den + (size_t)d * HEADS + h, ex);
    const float4* vp = (const float4*)(vr + (size_t)s * FDIM + h * DH);
    float* ap = agg + (size_t)d * FDIM + h * DH;
#pragma unroll
    for (int j = 0; j < 4; ++j) {
        float4 v = vp[j];
        atomicAdd(ap + 4 * j + 0, ex * v.x);
        atomicAdd(ap + 4 * j + 1, ex * v.y);
        atomicAdd(ap + 4 * j + 2, ex * v.z);
        atomicAdd(ap + 4 * j + 3, ex * v.w);
    }
}
// Pass 3: normalize, GELU, emit bf16 for output GEMM
__global__ void k_agg_gelu_bf16(const float* __restrict__ agg, const float* __restrict__ den,
                                unsigned short* __restrict__ out, int Nn) {
    int i = blockIdx.x * blockDim.x + threadIdx.x;
    if (i >= Nn * FDIM) return;
    int n = i >> 7, h = (i & 127) >> 4;
    float x = agg[i] / fmaxf(den[(size_t)n * HEADS + h], 1e-16f);
    out[i] = f2bf(gelu_exact(x));
}
// x = sig(skip)*out + (1-sig(skip))*x  (in place) and bf16 mirror
__global__ void k_skip_combine(const float* __restrict__ out, float* __restrict__ x,
                               unsigned short* __restrict__ xbf,
                               const float* __restrict__ skip, int n) {
    int i = blockIdx.x * blockDim.x + threadIdx.x;
    if (i >= n) return;
    float a = 1.0f / (1.0f + __expf(-skip[0]));
    float v = a * out[i] + (1.0f - a) * x[i];
    x[i]   = v;
    xbf[i] = f2bf(v);
}

// ---------------------------------------------------------------------------
// JK multihead self-attention over L=4 snapshots (tiny: 256 (n,h) pairs)
// qkv: [4,32,384] (q|k|v per row), o: [4,32,128]
// ---------------------------------------------------------------------------
__global__ void k_jk_attn(const float* __restrict__ qkv, float* __restrict__ o) {
    int t = blockIdx.x * blockDim.x + threadIdx.x;
    if (t >= NV * HEADS) return;
    int n = t >> 3, h = t & 7;
    float att[4][4];
    for (int l = 0; l < 4; ++l) {
        float mxv = -1e30f;
        const float* qp = qkv + (size_t)(l * NV + n) * 384 + h * DH;
        for (int m = 0; m < 4; ++m) {
            const float* kp = qkv + (size_t)(m * NV + n) * 384 + FDIM + h * DH;
            float s = 0.0f;
#pragma unroll
            for (int d = 0; d < DH; ++d) s += qp[d] * kp[d];
            s *= SCALE_ATT;
            att[l][m] = s;
            mxv = fmaxf(mxv, s);
        }
        float dsum = 0.0f;
        for (int m = 0; m < 4; ++m) { att[l][m] = __expf(att[l][m] - mxv); dsum += att[l][m]; }
        for (int m = 0; m < 4; ++m) att[l][m] /= dsum;
    }
    for (int l = 0; l < 4; ++l)
        for (int d = 0; d < DH; ++d) {
            float acc = 0.0f;
            for (int m = 0; m < 4; ++m)
                acc += att[l][m] * qkv[(size_t)(m * NV + n) * 384 + 256 + h * DH + d];
            o[(size_t)(l * NV + n) * FDIM + h * DH + d] = acc;
        }
}
__global__ void k_mean4(float* __restrict__ dst, const float* __restrict__ src) {
    int i = blockIdx.x * blockDim.x + threadIdx.x;
    if (i >= NV * FDIM) return;
    dst[i] = 0.25f * (src[i] + src[i + 4096] + src[i + 8192] + src[i + 12288]);
}

// ---------------------------------------------------------------------------
// Host orchestration
// ---------------------------------------------------------------------------
static inline unsigned grid1(long long n) { return (unsigned)((n + 255) / 256); }
#define L1(kern, n, ...) kern<<<grid1((long long)(n)), 256, 0, stream>>>(__VA_ARGS__)

extern "C" void kernel_launch(void* const* d_in, const int* in_sizes, int n_in,
                              void* d_out, int out_size, void* d_ws, size_t ws_size,
                              hipStream_t stream) {
    (void)in_sizes; (void)n_in; (void)out_size; (void)ws_size;
    const float* x_inst    = (const float*)d_in[0];
    const float* x_virtual = (const float*)d_in[1];
    const int*   ei_ii     = (const int*)d_in[2];     // [2,EII]: src row0, dst row1
    const int*   src_iv    = (const int*)d_in[3];
    const int*   dst_iv    = (const int*)d_in[4];
    const int*   src_vi    = (const int*)d_in[5];
    const int*   dst_vi    = (const int*)d_in[6];
    const float* Wkqv = (const float*)d_in[7];
    const float* bkqv = (const float*)d_in[8];
    const float* Wout = (const float*)d_in[9];
    const float* bout = (const float*)d_in[10];
    const float* skip = (const float*)d_in[11];
    const float* a_rel = (const float*)d_in[12];
    const float* m_rel = (const float*)d_in[13];
    const float* p_rel = (const float*)d_in[14];
    const float* Wjk_in  = (const float*)d_in[15];
    const float* bjk_in  = (const float*)d_in[16];
    const float* Wjk_out = (const float*)d_in[17];
    const float* bjk_out = (const float*)d_in[18];
    const float* Wm0 = (const float*)d_in[19];  const float* bm0 = (const float*)d_in[20];
    const float* Wm1 = (const float*)d_in[21];  const float* bm1 = (const float*)d_in[22];
    const float* Wm2 = (const float*)d_in[23];  const float* bm2 = (const float*)d_in[24];

    // ---- bump allocator over workspace ----
    char* base = (char*)d_ws;
    size_t off = 0;
    auto alloc = [&](size_t bytes) -> void* {
        off = (off + 255) & ~(size_t)255;
        void* p = base + off;
        off += bytes;
        return p;
    };
    const int Nn[2] = { NI, NV };

    float*          xf[2];   unsigned short* xbf[2];
    float*          kqv[2];  unsigned short* kbf[2]; unsigned short* vbf[2]; float* qf[2];
    float*          kr[3];   float* vr[3];
    unsigned short* gb[2];   float* ob[2];
    unsigned*       mxb[2];  float* den[2];  float* agg[2];

    for (int t = 0; t < 2; ++t) {
        xf[t]  = (float*)alloc((size_t)Nn[t] * FDIM * 4);
        xbf[t] = (unsigned short*)alloc((size_t)Nn[t] * FDIM * 2);
        kqv[t] = (float*)alloc((size_t)Nn[t] * 384 * 4);
        kbf[t] = (unsigned short*)alloc((size_t)Nn[t] * FDIM * 2);
        vbf[t] = (unsigned short*)alloc((size_t)Nn[t] * FDIM * 2);
        qf[t]  = (float*)alloc((size_t)Nn[t] * FDIM * 4);
        gb[t]  = (unsigned short*)alloc((size_t)Nn[t] * FDIM * 2);
        ob[t]  = (float*)alloc((size_t)Nn[t] * FDIM * 4);
        mxb[t] = (unsigned*)alloc((size_t)Nn[t] * HEADS * 4);
        den[t] = (float*)alloc((size_t)Nn[t] * HEADS * 4);
        agg[t] = (float*)alloc((size_t)Nn[t] * FDIM * 4);
    }
    const int relN[3] = { NI, NI, NV };        // src-node count per relation
    for (int r = 0; r < 3; ++r) {
        kr[r] = (float*)alloc((size_t)relN[r] * FDIM * 4);
        vr[r] = (float*)alloc((size_t)relN[r] * FDIM * 4);
    }
    float* sc0 = (float*)alloc((size_t)EII * HEADS * 4);
    float* sc1 = (float*)alloc((size_t)EIV * HEADS * 4);
    float* sc2 = (float*)alloc((size_t)EVI * HEADS * 4);

    unsigned short* w_kqv = (unsigned short*)alloc(128 * 384 * 2);
    unsigned short* w_out = (unsigned short*)alloc(128 * 128 * 2);
    unsigned short* bd_a  = (unsigned short*)alloc(128 * 128 * 2);
    unsigned short* bd_m  = (unsigned short*)alloc(128 * 128 * 2);

    float*          outs   = (float*)alloc(NLAYER * NV * FDIM * 4);
    unsigned short* jkx_bf = (unsigned short*)alloc(NLAYER * NV * FDIM * 2);
    unsigned short* w_jin  = (unsigned short*)alloc(128 * 384 * 2);
    float*          jkqkv  = (float*)alloc(128 * 384 * 4);
    float*          jko    = (float*)alloc(128 * 128 * 4);
    unsigned short* jko_bf = (unsigned short*)alloc(128 * 128 * 2);
    unsigned short* w_jout = (unsigned short*)alloc(128 * 128 * 2);
    float*          jo     = (float*)alloc(128 * 128 * 4);
    float*          mo     = (float*)alloc(NV * FDIM * 4);
    unsigned short* mo_bf  = (unsigned short*)alloc(NV * FDIM * 2);
    unsigned short* w_m0   = (unsigned short*)alloc(128 * 64 * 2);
    float*          h0     = (float*)alloc(NV * 64 * 4);
    unsigned short* h0_bf  = (unsigned short*)alloc(NV * 64 * 2);
    unsigned short* w_m1   = (unsigned short*)alloc(64 * 32 * 2);
    float*          h1     = (float*)alloc(NV * 32 * 4);
    unsigned short* h1_bf  = (unsigned short*)alloc(NV * 32 * 2);
    unsigned short* w_m2   = (unsigned short*)alloc(32 * 16 * 2);

    auto gemm = [&](const unsigned short* A, const unsigned short* Bp, const float* bias,
                    float* C, int N, int K, int M) {
        long long thr = (long long)(N >> 4) * (M >> 4) * 32;
        if (K == 128)
            k_gemm_bf16_t<128><<<grid1(thr), 256, 0, stream>>>(A, Bp, bias, C, N, M);
        else if (K == 64)
            k_gemm_bf16_t<64><<<grid1(thr), 256, 0, stream>>>(A, Bp, bias, C, N, M);
        else
            k_gemm_bf16_t<32><<<grid1(thr), 256, 0, stream>>>(A, Bp, bias, C, N, M);
    };

    // ---- init node features (never mutate inputs) ----
    L1(k_copy_f32,  NI * FDIM, xf[0], x_inst, NI * FDIM);
    L1(k_copy_f32,  NV * FDIM, xf[1], x_virtual, NV * FDIM);
    L1(k_f32_to_bf16, NI * FDIM, xbf[0], x_inst, NI * FDIM);
    L1(k_f32_to_bf16, NV * FDIM, xbf[1], x_virtual, NV * FDIM);

    const unsigned NEG_INF_KEY = 0x007FFFFFu;  // fkey(-inf)
    const int rel_src_type[3] = { 0, 0, 1 };

    for (int l = 0; l < NLAYER; ++l) {
        // KQV projections (WMMA) + slices
        for (int t = 0; t < 2; ++t) {
            L1(k_wconv_pack, 128 * 384, w_kqv, Wkqv + (size_t)(l * 2 + t) * 128 * 384,
               128, 384, 0);
            gemm(xbf[t], w_kqv, bkqv + (size_t)(l * 2 + t) * 384, kqv[t], Nn[t], 128, 384);
            L1(k_slice_bf16, Nn[t] * FDIM, kbf[t], kqv[t], Nn[t], FDIM, 384, 0);
            L1(k_slice_f32,  Nn[t] * FDIM, qf[t],  kqv[t], Nn[t], FDIM, 384, 128);
            L1(k_slice_bf16, Nn[t] * FDIM, vbf[t], kqv[t], Nn[t], FDIM, 384, 256);
        }
        // Relation transforms as block-diagonal 128x128 WMMA GEMMs
        for (int r = 0; r < 3; ++r) {
            int st = rel_src_type[r];
            L1(k_wblockdiag_pack, FDIM * FDIM, bd_a, a_rel + (size_t)(l * 3 + r) * 2048);
            L1(k_wblockdiag_pack, FDIM * FDIM, bd_m, m_rel + (size_t)(l * 3 + r) * 2048);
            gemm(kbf[st], bd_a, nullptr, kr[r], Nn[st], 128, 128);
            gemm(vbf[st], bd_m, nullptr, vr[r], Nn[st], 128, 128);
        }
        // Joint segment softmax + aggregation; dst type 0 <- rel 0,2 ; dst type 1 <- rel 1
        for (int t = 0; t < 2; ++t) {
            L1(k_fill_u32, Nn[t] * HEADS, mxb[t], NEG_INF_KEY, Nn[t] * HEADS);
            L1(k_fill_u32, Nn[t] * HEADS, (unsigned*)den[t], 0u, Nn[t] * HEADS);
            L1(k_fill_u32, Nn[t] * FDIM,  (unsigned*)agg[t], 0u, Nn[t] * FDIM);
        }
        const float* pr = p_rel + (size_t)l * 3 * HEADS;
        L1(k_edge_score, (long long)EII * HEADS, qf[0], kr[0], ei_ii, ei_ii + EII,
           pr + 0 * HEADS, EII, sc0, mxb[0]);
        L1(k_edge_score, (long long)EVI * HEADS, qf[0], kr[2], src_vi, dst_vi,
           pr + 2 * HEADS, EVI, sc2, mxb[0]);
        L1(k_edge_score, (long long)EIV * HEADS, qf[1], kr[1], src_iv, dst_iv,
           pr + 1 * HEADS, EIV, sc1, mxb[1]);
        L1(k_edge_accum, (long long)EII * HEADS, sc0, mxb[0], ei_ii, ei_ii + EII,
           vr[0], den[0], agg[0], EII);
        L1(k_edge_accum, (long long)EVI * HEADS, sc2, mxb[0], src_vi, dst_vi,
           vr[2], den[0], agg[0], EVI);
        L1(k_edge_accum, (long long)EIV * HEADS, sc1, mxb[1], src_iv, dst_iv,
           vr[1], den[1], agg[1], EIV);
        // normalize + GELU + output projection + gated skip
        for (int t = 0; t < 2; ++t) {
            L1(k_agg_gelu_bf16, Nn[t] * FDIM, agg[t], den[t], gb[t], Nn[t]);
            L1(k_wconv_pack, 128 * 128, w_out, Wout + (size_t)(l * 2 + t) * 128 * 128,
               128, 128, 0);
            gemm(gb[t], w_out, bout + (size_t)(l * 2 + t) * 128, ob[t], Nn[t], 128, 128);
            L1(k_skip_combine, Nn[t] * FDIM, ob[t], xf[t], xbf[t],
               skip + (size_t)(l * 2 + t), Nn[t] * FDIM);
        }
        // snapshot virtual nodes for JK
        L1(k_copy_f32, NV * FDIM, outs + (size_t)l * NV * FDIM, xf[1], NV * FDIM);
    }

    // ---- JK attention over [L=4, Nv=32, F=128] ----
    L1(k_f32_to_bf16, NLAYER * NV * FDIM, jkx_bf, outs, NLAYER * NV * FDIM);
    L1(k_wconv_pack, 128 * 384, w_jin, Wjk_in, 128, 384, 1);   // [384,128] -> packed
    gemm(jkx_bf, w_jin, bjk_in, jkqkv, NLAYER * NV, 128, 384);
    k_jk_attn<<<1, 256, 0, stream>>>(jkqkv, jko);
    L1(k_f32_to_bf16, 128 * 128, jko_bf, jko, 128 * 128);
    L1(k_wconv_pack, 128 * 128, w_jout, Wjk_out, 128, 128, 1);
    gemm(jko_bf, w_jout, bjk_out, jo, NLAYER * NV, 128, 128);
    L1(k_mean4, NV * FDIM, mo, jo);

    // ---- MLP head -> d_out [32,16] ----
    L1(k_f32_to_bf16, NV * FDIM, mo_bf, mo, NV * FDIM);
    L1(k_wconv_pack, 128 * 64, w_m0, Wm0, 128, 64, 1);
    gemm(mo_bf, w_m0, bm0, h0, NV, 128, 64);
    L1(k_gelu_bf16, NV * 64, h0_bf, h0, NV * 64);
    L1(k_wconv_pack, 64 * 32, w_m1, Wm1, 64, 32, 1);
    gemm(h0_bf, w_m1, bm1, h1, NV, 64, 32);
    L1(k_gelu_bf16, NV * 32, h1_bf, h1, NV * 32);
    L1(k_wconv_pack, 32 * 16, w_m2, Wm2, 32, 16, 1);
    gemm(h1_bf, w_m2, bm2, (float*)d_out, NV, 32, 16);
}
